// EBSDExperiment_56925496541961
// MI455X (gfx1250) — compile-verified, matrix operand order checked by reading, not run on hardware
//
#include <hip/hip_runtime.h>
#include <hip/hip_bf16.h>
#include <stdint.h>

// ---------------- problem constants (fixed by setup_inputs) ----------------
#define NQ    16384      // queries (patterns)
#define NPIX  3600       // pixels
#define KP    3616       // pixels padded to multiple of 32
#define ND    131072     // dictionary entries
#define NC    500        // PCA components
#define NCP   512        // components padded to multiple of 32
#define TOPK  8

typedef __bf16 bf16;
typedef __attribute__((ext_vector_type(16))) __bf16 v16bf;
typedef __attribute__((ext_vector_type(8)))  float  v8f;
typedef __attribute__((ext_vector_type(4)))  unsigned int u32x4;

union FragU { u32x4 q[2]; v16bf v; };

__device__ __forceinline__ v8f zero8() {
  v8f z;
#pragma unroll
  for (int i = 0; i < 8; ++i) z[i] = 0.0f;
  return z;
}

// Load a 16x32 bf16 A/B fragment from row-major [16 rows x ld] storage.
// ISA layout: lanes 0-15 hold row=lane, K 0..7 & 16..23; lanes 16-31 hold
// row=lane-16, K 8..15 & 24..31  ->  two 16B chunks at +ko and +ko+16 elems.
__device__ __forceinline__ v16bf load_frag(const bf16* base, int ld, int lane) {
  const int row = lane & 15;
  const int ko  = (lane >> 4) << 3;          // 0 or 8
  const u32x4* p = reinterpret_cast<const u32x4*>(base + row * ld + ko);
  FragU u;
  u.q[0] = p[0];        // elems [ko, ko+8)
  u.q[1] = p[2];        // elems [ko+16, ko+24)
  return u.v;
}

// ---------------- 1. per-row mean of X ----------------
__global__ __launch_bounds__(256) void ebsd_mean_kernel(const float* __restrict__ X,
                                                        float* __restrict__ means) {
  const int row  = blockIdx.x * 8 + (threadIdx.x >> 5);
  const int lane = threadIdx.x & 31;
  const float* xr = X + (size_t)row * NPIX;
  float s = 0.0f;
  for (int c = lane; c < NPIX; c += 32) s += xr[c];
#pragma unroll
  for (int m = 16; m; m >>= 1) s += __shfl_xor(s, m, 32);
  if (lane == 0) means[row] = s * (1.0f / (float)NPIX);
}

// ---------------- 2. column sums of P (for mean folding) ----------------
__global__ __launch_bounds__(256) void ebsd_colsum_kernel(const float* __restrict__ P,
                                                          float* __restrict__ svec) {
  const int n = blockIdx.x * 256 + threadIdx.x;
  if (n >= NCP) return;
  float s = 0.0f;
  if (n < NC)
    for (int k = 0; k < NPIX; ++k) s += P[(size_t)k * NC + n];
  svec[n] = s;
}

// ---------------- 3. P -> PT bf16 [NCP][KP], zero padded ----------------
__global__ __launch_bounds__(256) void ebsd_pt_kernel(const float* __restrict__ P,
                                                      bf16* __restrict__ PT) {
  const int idx = blockIdx.x * 256 + threadIdx.x;     // over NCP*KP
  const int n = idx / KP;
  const int k = idx - n * KP;
  float v = (n < NC && k < NPIX) ? P[(size_t)k * NC + n] : 0.0f;
  PT[(size_t)n * KP + k] = (bf16)v;
}

// ---------------- 4. dict -> bf16 [ND][NCP] + squared norms ----------------
__global__ __launch_bounds__(256) void ebsd_dictprep_kernel(const float* __restrict__ D,
                                                            bf16* __restrict__ Dbf,
                                                            float* __restrict__ dnorm) {
  const int row  = blockIdx.x * 8 + (threadIdx.x >> 5);
  const int lane = threadIdx.x & 31;
  const float* dr = D + (size_t)row * NC;
  bf16* qr = Dbf + (size_t)row * NCP;
  float s = 0.0f;
  for (int c = lane; c < NCP; c += 32) {
    float v = (c < NC) ? dr[c] : 0.0f;
    qr[c] = (bf16)v;
    s += v * v;
  }
#pragma unroll
  for (int m = 16; m; m >>= 1) s += __shfl_xor(s, m, 32);
  if (lane == 0) dnorm[row] = s;
}

// ---------------- 5. Q = X@P - mean ox colsum(P), bf16 out ----------------
// block: 256 thr (8 waves). tile: 128 (M) x 128 (N). wave w = M strip w*16.
__global__ __launch_bounds__(256) void ebsd_gemm1_kernel(const float* __restrict__ X,
                                                         const bf16* __restrict__ PT,
                                                         const float* __restrict__ means,
                                                         const float* __restrict__ svec,
                                                         bf16* __restrict__ Qbf) {
  __shared__ bf16 As[128 * 32];
  __shared__ bf16 Bs[128 * 32];
  const int t = threadIdx.x, w = t >> 5, lane = t & 31;
  const int m0 = blockIdx.x * 128;
  const int n0 = blockIdx.y * 128;

  v8f acc[8];
#pragma unroll
  for (int i = 0; i < 8; ++i) acc[i] = zero8();

  const int sr = t >> 1, sc = (t & 1) * 16;   // staging row/col (vector path)

  for (int k0 = 0; k0 < KP; k0 += 32) {
    // stage A: fp32 -> bf16 (needs convert + tail guard; scalar)
#pragma unroll
    for (int i = 0; i < 16; ++i) {
      const int idx = i * 256 + t;
      const int r = idx >> 5, c = idx & 31;
      const int kc = k0 + c;
      float v = (kc < NPIX) ? X[(size_t)(m0 + r) * NPIX + kc] : 0.0f;
      As[idx] = (bf16)v;
    }
    // stage B: straight bf16 copy, 32B per thread
    {
      const u32x4* src = (const u32x4*)(PT + (size_t)(n0 + sr) * KP + k0 + sc);
      u32x4* dst = (u32x4*)(Bs + sr * 32 + sc);
      dst[0] = src[0];
      dst[1] = src[1];
    }
    __syncthreads();
    v16bf a = load_frag(As + (w * 16) * 32, 32, lane);
#pragma unroll
    for (int nt = 0; nt < 8; ++nt) {
      v16bf b = load_frag(Bs + (nt * 16) * 32, 32, lane);
      acc[nt] = __builtin_amdgcn_wmma_f32_16x16x32_bf16(
          false, a, false, b, (short)0, acc[nt], false, false);
    }
    __syncthreads();
  }

  // epilogue: subtract mean[m]*colsum[n], convert, store
  const int hi = lane >> 4;
  float mv[8];
#pragma unroll
  for (int v = 0; v < 8; ++v) mv[v] = means[m0 + w * 16 + v + hi * 8];
#pragma unroll
  for (int nt = 0; nt < 8; ++nt) {
    const int col = n0 + nt * 16 + (lane & 15);
    const float sv = svec[col];
#pragma unroll
    for (int v = 0; v < 8; ++v) {
      const int row = m0 + w * 16 + v + hi * 8;
      Qbf[(size_t)row * NCP + col] = (bf16)(acc[nt][v] - mv[v] * sv);
    }
  }
}

// ---------------- 6. fused distance GEMM + running top-8 ----------------
// block: 256 thr (8 waves). 64-query tile resident in LDS; dict streamed in
// 128-entry chunks via double-buffered GLOBAL_LOAD_ASYNC_TO_LDS_B128.
// score = dnorm[n] - 2*q.d  (q_norm dropped: rank-invariant per query row)
#define SC_LD 132   // padded score stride (bank-conflict dodge)
#define SMEM_Q   (64 * NCP * 2)                  // 65536
#define SMEM_D1  (128 * 32 * 2)                  // 8192 per buffer
#define SMEM_D2  (2 * SMEM_D1)                   // 16384 double-buffered
#define SMEM_SC  (64 * SC_LD * 4)                // 33792
#define KNN_SMEM (SMEM_Q + SMEM_D2 + SMEM_SC)    // 115712

__global__ __launch_bounds__(256) void ebsd_knn_kernel(const bf16* __restrict__ Qbf,
                                                       const bf16* __restrict__ Dbf,
                                                       const float* __restrict__ dnorm,
                                                       int* __restrict__ out) {
  extern __shared__ char smem[];   // no static LDS in this kernel -> dynamic base = offset 0
  bf16*  Qs = (bf16*)smem;
  bf16*  Ds = (bf16*)(smem + SMEM_Q);              // two 8KB dict-tile buffers
  float* Sc = (float*)(smem + SMEM_Q + SMEM_D2);

  const int t = threadIdx.x, w = t >> 5, lane = t & 31;
  const int m0 = blockIdx.x * 64;

  // stage the whole 64x512 bf16 query tile (contiguous rows)
  {
    const u32x4* src = (const u32x4*)(Qbf + (size_t)m0 * NCP);
    u32x4* dst = (u32x4*)Qs;
#pragma unroll
    for (int i = 0; i < 16; ++i) dst[i * 256 + t] = src[i * 256 + t];
  }
  __syncthreads();

  const int ms = (w & 3) * 16;      // query strip for this wave
  const int nh = (w >> 2) * 64;     // dict half for this wave

  // async-copy staging coordinates: 2 threads per dict row, 32B each
  const int sr = t >> 1, sc2 = (t & 1) * 16;
  const unsigned ldsA = (unsigned)(SMEM_Q + (sr * 32 + sc2) * 2);          // buf0 byte offset
  const unsigned long long growA =
      (unsigned long long)(uintptr_t)Dbf + ((unsigned long long)sr * NCP + sc2) * 2;

  // per-thread running top-8 (4 threads per query, 32 cols each)
  const int q = t & 63, part = t >> 6;
  float bd[8]; int bi[8];
#pragma unroll
  for (int i = 0; i < 8; ++i) { bd[i] = 3.4e38f; bi[i] = 0; }
  float worst = 3.4e38f; int wslot = 0;

  for (int n0 = 0; n0 < ND; n0 += 128) {
    v8f acc[4];
#pragma unroll
    for (int i = 0; i < 4; ++i) acc[i] = zero8();

    // prologue: async-copy tile 0 of this chunk into buffer 0
    {
      unsigned long long ga = growA + ((unsigned long long)n0 * NCP) * 2;
      asm volatile(
          "global_load_async_to_lds_b128 %0, %1, off\n\t"
          "global_load_async_to_lds_b128 %2, %3, off"
          :: "v"(ldsA), "v"(ga), "v"(ldsA + 16u), "v"(ga + 16ull)
          : "memory");
    }

    for (int ks = 0; ks < NCP / 32; ++ks) {
      const int buf = ks & 1;
      if (ks + 1 < NCP / 32) {
        // async-copy next tile into the other buffer, then wait for current
        unsigned ld = ldsA + (unsigned)((buf ^ 1) * SMEM_D1);
        unsigned long long ga =
            growA + ((unsigned long long)n0 * NCP + (unsigned long long)(ks + 1) * 32) * 2;
        asm volatile(
            "global_load_async_to_lds_b128 %0, %1, off\n\t"
            "global_load_async_to_lds_b128 %2, %3, off"
            :: "v"(ld), "v"(ga), "v"(ld + 16u), "v"(ga + 16ull)
            : "memory");
        asm volatile("s_wait_asynccnt 0x2" ::: "memory");   // older 2 (current buf) done
      } else {
        asm volatile("s_wait_asynccnt 0x0" ::: "memory");
      }
      __syncthreads();   // current buffer visible to all waves

      v16bf a = load_frag(Qs + (size_t)ms * NCP + ks * 32, NCP, lane);
      const bf16* Db = Ds + buf * (128 * 32);
#pragma unroll
      for (int nt = 0; nt < 4; ++nt) {
        v16bf b = load_frag(Db + (nh + nt * 16) * 32, 32, lane);
        acc[nt] = __builtin_amdgcn_wmma_f32_16x16x32_bf16(
            false, a, false, b, (short)0, acc[nt], false, false);
      }
      __syncthreads();   // frag reads done before this buffer is re-filled
    }

    // scores -> LDS
    const int hi = lane >> 4;
#pragma unroll
    for (int nt = 0; nt < 4; ++nt) {
      const int nl = nh + nt * 16 + (lane & 15);
      const float dn = dnorm[n0 + nl];
#pragma unroll
      for (int v = 0; v < 8; ++v) {
        const int ql = ms + v + hi * 8;
        Sc[ql * SC_LD + nl] = dn - 2.0f * acc[nt][v];
      }
    }
    __syncthreads();

    // running top-8 scan (replace-current-worst policy)
    const float* sr2 = Sc + q * SC_LD + part * 32;
#pragma unroll
    for (int j = 0; j < 32; ++j) {
      const float d = sr2[j];
      if (d < worst) {
        bd[wslot] = d; bi[wslot] = n0 + part * 32 + j;
        worst = bd[0]; wslot = 0;
#pragma unroll
        for (int i = 1; i < 8; ++i)
          if (bd[i] > worst) { worst = bd[i]; wslot = i; }
      }
    }
    __syncthreads();
  }

  // merge the 4 partial top-8 lists per query (reuse Qs region)
  float* md = (float*)smem;             // 64*32 floats = 8 KB
  int*   mi = (int*)(smem + 64 * 32 * 4);
#pragma unroll
  for (int i = 0; i < 8; ++i) {
    md[q * 32 + part * 8 + i] = bd[i];
    mi[q * 32 + part * 8 + i] = bi[i];
  }
  __syncthreads();
  if (t < 64) {
    float* rd = md + t * 32;
    int*   ri = mi + t * 32;
#pragma unroll
    for (int r = 0; r < TOPK; ++r) {
      float best = rd[0]; int bp = 0;
      for (int j = 1; j < 32; ++j)
        if (rd[j] < best) { best = rd[j]; bp = j; }
      out[(size_t)(m0 + t) * TOPK + r] = ri[bp];
      rd[bp] = 3.4e38f;
    }
  }
}

// ---------------- launcher ----------------
extern "C" void kernel_launch(void* const* d_in, const int* in_sizes, int n_in,
                              void* d_out, int out_size, void* d_ws, size_t ws_size,
                              hipStream_t stream) {
  const float* X    = (const float*)d_in[0];   // [NQ, NPIX]
  const float* P    = (const float*)d_in[1];   // [NPIX, NC]
  const float* Dict = (const float*)d_in[2];   // [ND, NC]
  int* out = (int*)d_out;                      // [NQ, TOPK] int32 indices

  char* ws = (char*)d_ws;
  // workspace layout (256B-aligned blocks)
  bf16*  Dbf   = (bf16*)(ws + 0);                       // ND*NCP*2   = 134217728
  bf16*  Qbf   = (bf16*)(ws + 134217728);               // NQ*NCP*2   =  16777216
  bf16*  PT    = (bf16*)(ws + 150994944);               // NCP*KP*2   =   3702784
  float* dnorm = (float*)(ws + 154697728);              // ND*4       =    524288
  float* means = (float*)(ws + 155222016);              // NQ*4       =     65536
  float* svec  = (float*)(ws + 155287552);              // NCP*4      =      2048

  ebsd_mean_kernel<<<NQ / 8, 256, 0, stream>>>(X, means);
  ebsd_colsum_kernel<<<2, 256, 0, stream>>>(P, svec);
  ebsd_pt_kernel<<<(NCP * KP) / 256, 256, 0, stream>>>(P, PT);
  ebsd_dictprep_kernel<<<ND / 8, 256, 0, stream>>>(Dict, Dbf, dnorm);
  ebsd_gemm1_kernel<<<dim3(NQ / 128, NCP / 128), 256, 0, stream>>>(X, PT, means, svec, Qbf);
  ebsd_knn_kernel<<<NQ / 64, 256, KNN_SMEM, stream>>>(Qbf, Dbf, dnorm, out);
}
